// TimeMix_78065325572197
// MI455X (gfx1250) — matrix operand order repaired and checked by reading, not compiled, not run on hardware
//
#include <hip/hip_runtime.h>
#include <hip/hip_bf16.h>

#define DEV __device__ __forceinline__

typedef __attribute__((ext_vector_type(16))) __bf16 v16bf;
typedef __attribute__((ext_vector_type(8)))  __bf16 v8bf;
typedef __attribute__((ext_vector_type(4)))  __bf16 v4bf;
typedef __attribute__((ext_vector_type(8)))  float  v8f;

constexpr int Dd = 2048;
constexpr int Tt = 2048;
constexpr int Bb = 8;
constexpr int Mm = Bb * Tt;   // 16384 rows

union ABf { v16bf v; v8bf h[2]; };

// ---- fragment loaders (layouts per CDNA5 ISA 7.12.2) -----------------------

// B matrix 32x16 (KxN), B[k][n] = W[n*D + k]. Lane holds column n = n0+(lane&15),
// elements 0..7 = K klo..klo+7, 8..15 = K klo+16..klo+23, klo = 0 / 8 (lanes 16-31).
DEV v16bf load_b_frag(const __bf16* __restrict__ W, int n0, int lane, int k0) {
    int n  = n0 + (lane & 15);
    int kb = (lane & 16) ? 8 : 0;
    const __bf16* p = W + (size_t)n * Dd + k0 + kb;
    ABf u;
    u.h[0] = *(const v8bf*)(p);
    u.h[1] = *(const v8bf*)(p + 16);
    return u.v;
}

// A matrix 16x32 (MxK). Lane holds row m = m0+(lane&15); elements 0..7 = K klo..klo+7,
// elements 8..15 = K klo+16..klo+23, with klo = 0 (lanes 0-15) or 8 (lanes 16-31).
DEV v16bf load_a_frag(const __bf16* __restrict__ X, int m0, int lane, int k0) {
    int r  = m0 + (lane & 15);
    int kl = (lane & 16) ? 8 : 0;
    const __bf16* p = X + (size_t)r * Dd + k0 + kl;
    ABf u;
    u.h[0] = *(const v8bf*)(p);
    u.h[1] = *(const v8bf*)(p + 16);
    return u.v;
}

// Same but with RWKV time-shift folded in: row m reads x row m-1; t==0 rows are zero.
DEV v16bf load_a_frag_shift(const __bf16* __restrict__ X, int m0, int lane, int k0) {
    int r  = m0 + (lane & 15);
    v8bf z = {};
    ABf u; u.h[0] = z; u.h[1] = z;
    if ((r & (Tt - 1)) != 0) {
        int kl = (lane & 16) ? 8 : 0;
        const __bf16* p = X + (size_t)(r - 1) * Dd + k0 + kl;
        u.h[0] = *(const v8bf*)(p);
        u.h[1] = *(const v8bf*)(p + 16);
    }
    return u.v;
}

DEV v8f wmma_bf16(v16bf a, v16bf b, v8f c) {
    // (neg_a, A, neg_b, B, c_mod, C, reuse_a, reuse_b)
    return __builtin_amdgcn_wmma_f32_16x16x32_bf16(false, a, false, b, (short)0, c,
                                                   false, false);
}

// ---- f32 -> bf16 conversion (vectorized) -----------------------------------

__global__ __launch_bounds__(256) void cvt_bf16_kernel(const float* __restrict__ s,
                                                       __bf16* __restrict__ d, int n4) {
    int i = blockIdx.x * blockDim.x + threadIdx.x;
    if (i < n4) {
        float4 f = ((const float4*)s)[i];
        v4bf o;
        o[0] = (__bf16)f.x; o[1] = (__bf16)f.y; o[2] = (__bf16)f.z; o[3] = (__bf16)f.w;
        ((v4bf*)d)[i] = o;
    }
}

// ---- fused k/v GEMM: ek = exp(shifted@Wk^T), a = ek * (shifted@Wv^T) -------
// Block = 8 waves; block tile 128x128; wave tile 64(M) x 32(N) = 4x2 WMMA tiles.

__global__ __launch_bounds__(256) void kv_wmma_kernel(const __bf16* __restrict__ xb,
                                                      const __bf16* __restrict__ wkb,
                                                      const __bf16* __restrict__ wvb,
                                                      float* __restrict__ ekb,
                                                      float* __restrict__ ab) {
    const int lane = threadIdx.x & 31;
    const int wave = threadIdx.x >> 5;
    const int m0 = blockIdx.y * 128 + (wave >> 2) * 64;
    const int n0 = blockIdx.x * 128 + (wave & 3) * 32;

    v8f acck[4][2] = {};
    v8f accv[4][2] = {};

    for (int k0 = 0; k0 < Dd; k0 += 32) {
        v16bf afr[4];
#pragma unroll
        for (int mt = 0; mt < 4; ++mt)
            afr[mt] = load_a_frag_shift(xb, m0 + 16 * mt, lane, k0);
#pragma unroll
        for (int nt = 0; nt < 2; ++nt) {
            v16bf bk = load_b_frag(wkb, n0 + 16 * nt, lane, k0);
            v16bf bv = load_b_frag(wvb, n0 + 16 * nt, lane, k0);
#pragma unroll
            for (int mt = 0; mt < 4; ++mt) {
                acck[mt][nt] = wmma_bf16(afr[mt], bk, acck[mt][nt]);
                accv[mt][nt] = wmma_bf16(afr[mt], bv, accv[mt][nt]);
            }
        }
    }

    // C/D layout: n = n0 + (lane&15); m = m0 + j + (lane>=16 ? 8 : 0)
    const int nlane = lane & 15;
    const int mhalf = (lane & 16) ? 8 : 0;
#pragma unroll
    for (int mt = 0; mt < 4; ++mt)
#pragma unroll
        for (int nt = 0; nt < 2; ++nt)
#pragma unroll
            for (int j = 0; j < 8; ++j) {
                int m = m0 + 16 * mt + mhalf + j;
                int n = n0 + 16 * nt + nlane;
                size_t idx = (size_t)m * Dd + n;
                float e = __expf(acck[mt][nt][j]);
                ekb[idx] = e;
                ab[idx]  = e * accv[mt][nt][j];
            }
}

// ---- r GEMM: r = x @ Wr^T, written straight into d_out ---------------------

__global__ __launch_bounds__(256) void r_wmma_kernel(const __bf16* __restrict__ xb,
                                                     const __bf16* __restrict__ wrb,
                                                     float* __restrict__ out) {
    const int lane = threadIdx.x & 31;
    const int wave = threadIdx.x >> 5;
    const int m0 = blockIdx.y * 128 + (wave >> 2) * 64;
    const int n0 = blockIdx.x * 128 + (wave & 3) * 32;

    v8f acc[4][2] = {};

    for (int k0 = 0; k0 < Dd; k0 += 32) {
        v16bf afr[4];
#pragma unroll
        for (int mt = 0; mt < 4; ++mt)
            afr[mt] = load_a_frag(xb, m0 + 16 * mt, lane, k0);
#pragma unroll
        for (int nt = 0; nt < 2; ++nt) {
            v16bf br = load_b_frag(wrb, n0 + 16 * nt, lane, k0);
#pragma unroll
            for (int mt = 0; mt < 4; ++mt)
                acc[mt][nt] = wmma_bf16(afr[mt], br, acc[mt][nt]);
        }
    }

    const int nlane = lane & 15;
    const int mhalf = (lane & 16) ? 8 : 0;
#pragma unroll
    for (int mt = 0; mt < 4; ++mt)
#pragma unroll
        for (int nt = 0; nt < 2; ++nt)
#pragma unroll
            for (int j = 0; j < 8; ++j) {
                int m = m0 + 16 * mt + mhalf + j;
                int n = n0 + 16 * nt + nlane;
                out[(size_t)m * Dd + n] = acc[mt][nt][j];
            }
}

// ---- WKV recurrence along T; one lane per (b,d) channel, coalesced in d ----
// Reads r from d_out and overwrites with r * wkv (same element, same thread).

__global__ __launch_bounds__(256) void wkv_scan_kernel(const float* __restrict__ ekb,
                                                       const float* __restrict__ ab,
                                                       const float* __restrict__ td,
                                                       float* __restrict__ out) {
    int g = blockIdx.x * blockDim.x + threadIdx.x;   // 0 .. B*D-1
    int b = g / Dd;
    int d = g - b * Dd;
    float w = __expf(-__expf(td[d]));
    float num = 0.f, den = 0.f;
    size_t idx = (size_t)b * Tt * Dd + d;
    for (int t = 0; t < Tt; ++t, idx += Dd) {
        num = w * num + ab[idx];
        den = w * den + ekb[idx];
        float r = out[idx];
        out[idx] = r * (num / (den + 1e-8f));
    }
}

// ---- second output: shifted[:, -1] = x[:, T-2] -----------------------------

__global__ __launch_bounds__(256) void last_shift_kernel(const float* __restrict__ x,
                                                         float* __restrict__ out2) {
    int g = blockIdx.x * blockDim.x + threadIdx.x;   // 0 .. B*D-1
    int b = g / Dd;
    int d = g - b * Dd;
    out2[g] = x[((size_t)b * Tt + (Tt - 2)) * Dd + d];
}

// ---- launcher --------------------------------------------------------------

extern "C" void kernel_launch(void* const* d_in, const int* in_sizes, int n_in,
                              void* d_out, int out_size, void* d_ws, size_t ws_size,
                              hipStream_t stream) {
    (void)in_sizes; (void)n_in; (void)out_size; (void)ws_size;

    const float* x  = (const float*)d_in[0];
    const float* Wk = (const float*)d_in[1];
    const float* Wv = (const float*)d_in[2];
    const float* Wr = (const float*)d_in[3];
    const float* td = (const float*)d_in[4];
    float* out = (float*)d_out;

    char* ws = (char*)d_ws;
    __bf16* xb  = (__bf16*)ws;  ws += (size_t)Mm * Dd * 2;   //  64 MB
    __bf16* wkb = (__bf16*)ws;  ws += (size_t)Dd * Dd * 2;   //   8 MB
    __bf16* wvb = (__bf16*)ws;  ws += (size_t)Dd * Dd * 2;   //   8 MB
    __bf16* wrb = (__bf16*)ws;  ws += (size_t)Dd * Dd * 2;   //   8 MB
    float*  ekb = (float*)ws;   ws += (size_t)Mm * Dd * 4;   // 128 MB
    float*  ab  = (float*)ws;                                // 128 MB

    // 1) f32 -> bf16 conversions
    {
        int n4 = Mm * Dd / 4;
        cvt_bf16_kernel<<<n4 / 256, 256, 0, stream>>>(x, xb, n4);
        n4 = Dd * Dd / 4;
        cvt_bf16_kernel<<<n4 / 256, 256, 0, stream>>>(Wk, wkb, n4);
        cvt_bf16_kernel<<<n4 / 256, 256, 0, stream>>>(Wv, wvb, n4);
        cvt_bf16_kernel<<<n4 / 256, 256, 0, stream>>>(Wr, wrb, n4);
    }

    // 2) WMMA GEMMs
    dim3 grid(Dd / 128, Mm / 128);   // (16, 128)
    kv_wmma_kernel<<<grid, 256, 0, stream>>>(xb, wkb, wvb, ekb, ab);
    r_wmma_kernel<<<grid, 256, 0, stream>>>(xb, wrb, out);

    // 3) sequential WKV scan + epilogue (overwrites r in d_out with r*wkv)
    wkv_scan_kernel<<<(Bb * Dd) / 256, 256, 0, stream>>>(ekb, ab, td, out);

    // 4) second output: shifted[:, -1]
    last_shift_kernel<<<(Bb * Dd) / 256, 256, 0, stream>>>(x, out + (size_t)Mm * Dd);
}
